// InterpretableMultiHeadAttention_17798344475070
// MI455X (gfx1250) — matrix operand order repaired and checked by reading, not compiled, not run on hardware
//
#include <hip/hip_runtime.h>

#define DEV_INLINE __device__ __forceinline__

typedef __attribute__((ext_vector_type(16))) __bf16 bf16x16;
typedef __attribute__((ext_vector_type(8)))  float  floatx8;
typedef __attribute__((ext_vector_type(4)))  unsigned int uint32x4;
typedef __attribute__((ext_vector_type(8)))  int    int32x8;
typedef __attribute__((ext_vector_type(4)))  int    int32x4;

constexpr int B  = 16;
constexpr int S  = 1024;
constexpr int D  = 512;
constexpr int H  = 8;
constexpr int DH = 64;
constexpr float NEGBIG = -1e9f;

#if defined(__has_builtin)
#if __has_builtin(__builtin_amdgcn_tensor_load_to_lds) && \
    __has_builtin(__builtin_amdgcn_s_wait_tensorcnt)
#define HAVE_TDM 1
#endif
#endif

DEV_INLINE floatx8 wmma_bf16(const bf16x16& a, const bf16x16& b, const floatx8& c) {
  // D = A(16x32 bf16) * B(32x16 bf16) + C(16x16 f32)
  return __builtin_amdgcn_wmma_f32_16x16x32_bf16(false, a, false, b, (short)0, c,
                                                 false, false);
}

// ---------------------------------------------------------------------------
__global__ void zero_kernel(float* __restrict__ p, long n) {
  long i = (long)blockIdx.x * blockDim.x + threadIdx.x;
  if (i < n) p[i] = 0.0f;
}

// ---------------------------------------------------------------------------
// Projection GEMM: Y[b,h,s,:DH] = X[b,s,:D] (f32) * W[h,:D,:DH] (f32), bf16 out
// grid = (S/16, B*nHeads), block = 128 (4 waves, each owns one 16-wide DH tile)
// Weight tiles are staged TRANSPOSED in LDS (bf16) so every lane's B fragment
// is a contiguous 32B run (2x ds_load_b128) instead of 16 strided vmem dwords.
// ---------------------------------------------------------------------------
__global__ void proj_kernel(const float* __restrict__ X,
                            const float* __restrict__ W,
                            __bf16* __restrict__ Y,
                            int nHeads, long wHeadStride) {
  __shared__ __bf16 wt[64 * 32];          // W tile, transposed: wt[n*32 + k]
  const int lane = threadIdx.x & 31;
  const int wave = threadIdx.x >> 5;      // 0..3 -> which 16-col tile of DH
  const int s0   = blockIdx.x * 16;
  const int b    = blockIdx.y / nHeads;
  const int h    = blockIdx.y % nHeads;
  const int n0   = wave * 16;

  const int arow = lane & 15;
  const int ak   = (lane & 16) ? 16 : 0;
  const int bcol = lane & 15;
  const int bk   = (lane & 16) ? 16 : 0;

  const float* __restrict__ Xrow = X + ((long)b * S + s0 + arow) * D;
  const float* __restrict__ Wh   = W + (long)h * wHeadStride;

  floatx8 acc = {};
  for (int kk = 0; kk < D; kk += 32) {
    __builtin_prefetch(Xrow + kk + 64, 0, 0);   // global_prefetch_b8
    // ---- stage 32x64 weight tile, transposed, coalesced ----
    __syncthreads();
    for (int e = threadIdx.x; e < 32 * 64; e += 128) {
      const int k = e >> 6;       // 0..31
      const int n = e & 63;       // 0..63 (consecutive threads -> coalesced)
      wt[n * 32 + k] = (__bf16)Wh[(long)(kk + k) * DH + n];
    }
    __syncthreads();
    // ---- fragments ----
    bf16x16 a, bm;
#pragma unroll
    for (int i = 0; i < 8; ++i) {
      a[i]     = (__bf16)Xrow[kk + ak + i];        // K = kk+ak+0..7
      a[8 + i] = (__bf16)Xrow[kk + ak + 8 + i];    // K = kk+ak+8..15
    }
    const __bf16* __restrict__ wrow = wt + (n0 + bcol) * 32;
#pragma unroll
    for (int j = 0; j < 16; ++j) bm[j] = wrow[bk + j];   // contiguous 32B
    acc = wmma_bf16(a, bm, acc);
  }

  __bf16* __restrict__ Yb = Y + ((long)(b * nHeads + h) * S + s0) * DH;
#pragma unroll
  for (int r = 0; r < 8; ++r) {
    const int m = r + ((lane & 16) ? 8 : 0);
    Yb[(long)m * DH + n0 + (lane & 15)] = (__bf16)acc[r];
  }
}

// ---------------------------------------------------------------------------
// Fused attention: scores -> softmax -> attn write -> heads -> mean-accumulate
// grid = (S/16, B*H), block = 256 (8 waves)
// Dynamic LDS: 64KB f32 scores + 32KB bf16 P + 128KB bf16 vs-panel = 224KB
// (fits the 320KB WGP LDS). The vs panel is fetched by the Tensor Data Mover
// at kernel entry, overlapping the whole score/softmax phase; we only
// s_wait_tensorcnt right before the P*V GEMM.
// ---------------------------------------------------------------------------
__global__ void attn_kernel(const __bf16* __restrict__ qh,   // [B,H,S,DH]
                            const __bf16* __restrict__ kh,   // [B,H,S,DH]
                            const __bf16* __restrict__ vs,   // [B,S,DH]
                            const unsigned char* __restrict__ mask, // [B,S,S]
                            float* __restrict__ attn,        // [B,H,S,S]
                            float* __restrict__ hsum)        // [B,S,DH]
{
  extern __shared__ char smem[];
  float*  sc   = (float*)smem;                        // 16 x S f32   (64KB)
  __bf16* p_bf = (__bf16*)(smem + 16 * S * 4);        // 16 x S bf16  (32KB)
  __bf16* vs_l = (__bf16*)(smem + 16 * S * 4 + 16 * S * 2); // S x DH (128KB)

  const int lane = threadIdx.x & 31;
  const int wave = threadIdx.x >> 5;       // 0..7
  const int s0 = blockIdx.x * 16;
  const int b  = blockIdx.y >> 3;
  const int h  = blockIdx.y & 7;

  const int arow = lane & 15;
  const int ak   = (lane & 16) ? 16 : 0;
  const int bcol = lane & 15;
  const int bk   = (lane & 16) ? 16 : 0;
  const int cm   = (lane & 16) ? 8 : 0;
  const int cn   = lane & 15;

  const __bf16* __restrict__ vb = vs + (long)b * S * DH;

#ifdef HAVE_TDM
  // ---- kick off async TDM copy of the whole vs[b] panel into LDS ---------
  if (threadIdx.x < 32) {                  // wave 0 issues the DMA
    const unsigned lds_addr = (unsigned)(unsigned long long)(const void*)vs_l;
    const unsigned long long ga = (unsigned long long)(const void*)vb;
    uint32x4 g0;
    g0.x = 1u;                                           // count=1 (valid D#)
    g0.y = lds_addr;                                     // lds_addr[31:0]
    g0.z = (unsigned)(ga & 0xffffffffull);               // global_addr[31:0]
    g0.w = (unsigned)((ga >> 32) & 0x01ffffffull)        // global_addr[56:32]
         | (2u << 30);                                   // type=2 ("image")
    int32x8 g1;
    g1[0] = (1 << 16);                 // workgroup_mask=0, data_size=1 (2B)
    g1[1] = (DH & 0xffff) << 16;       // tensor_dim0[15:0] (bits 63:48)
    g1[2] = (S  & 0xffff) << 16;       // dim0 hi=0 | tensor_dim1[15:0]
    g1[3] = (DH & 0xffff) << 16;       // dim1 hi=0 | tile_dim0 = DH
    g1[4] = (S  & 0xffff);             // tile_dim1 = S, tile_dim2 = 0
    g1[5] = DH;                        // tensor_dim0_stride[31:0] = DH
    g1[6] = 0;                         // stride hi / dim1_stride (unused, 2D)
    g1[7] = 0;
    const int32x4 gz4 = {0, 0, 0, 0};
    const int32x8 gz8 = {0, 0, 0, 0, 0, 0, 0, 0};
    // 6-arg variant (clang-23 / therock-10.0 headers): g0,g1,g2,g3,g4,cpol
    __builtin_amdgcn_tensor_load_to_lds(g0, g1, gz4, gz4, gz8, 0);
  }
#endif

  // ---- Phase 1: 16 x S raw scores into LDS -------------------------------
  const __bf16* __restrict__ qrow = qh + ((long)(b*H + h) * S + s0 + arow) * DH;
  bf16x16 a0, a1;
#pragma unroll
  for (int i = 0; i < 8; ++i) {
    a0[i]     = qrow[ak + i];
    a0[8 + i] = qrow[8 + ak + i];
    a1[i]     = qrow[32 + ak + i];
    a1[8 + i] = qrow[40 + ak + i];
  }
  const __bf16* __restrict__ kbh = kh + (long)(b*H + h) * S * DH;
  const unsigned char* __restrict__ mrowbase = mask + (long)b * S * S;

  for (int j = 0; j < 8; ++j) {
    const int t0 = (wave * 8 + j) * 16;
    const __bf16* __restrict__ krow = kbh + (long)(t0 + bcol) * DH;
    bf16x16 b0, b1;
#pragma unroll
    for (int i = 0; i < 16; ++i) {       // contiguous 32B runs -> b128 loads
      b0[i] = krow[bk + i];
      b1[i] = krow[32 + bk + i];
    }
    floatx8 acc = {};
    acc = wmma_bf16(a0, b0, acc);
    acc = wmma_bf16(a1, b1, acc);
#pragma unroll
    for (int r = 0; r < 8; ++r) {
      const int m = r + cm;
      const float mval = (float)mrowbase[(long)(s0 + m) * S + t0 + cn];
      sc[m * S + t0 + cn] = acc[r] * 0.125f + (1.0f - mval) * NEGBIG;
    }
  }
  __syncthreads();

  // ---- Phase 2: softmax per row (16 lanes per row), stream attn out ------
  {
    const int r = threadIdx.x >> 4;        // 0..15
    const int c = threadIdx.x & 15;
    float* __restrict__ row = sc + r * S;
    float mx = -3.402823e38f;
    for (int jj = c; jj < S; jj += 16) mx = fmaxf(mx, row[jj]);
#pragma unroll
    for (int off = 8; off; off >>= 1) mx = fmaxf(mx, __shfl_xor(mx, off, 32));
    float sum = 0.0f;
    for (int jj = c; jj < S; jj += 16) {
      const float e = __expf(row[jj] - mx);
      row[jj] = e;
      sum += e;
    }
#pragma unroll
    for (int off = 8; off; off >>= 1) sum += __shfl_xor(sum, off, 32);
    const float inv = 1.0f / sum;
    float* __restrict__ arow_g = attn + ((long)(b*H + h) * S + s0 + r) * S;
    for (int jj = c; jj < S; jj += 16) {
      const float p = row[jj] * inv;
      p_bf[r * S + jj] = (__bf16)p;                 // bf16 P for the P*V GEMM
      __builtin_nontemporal_store(p, arow_g + jj);  // pure streaming output
    }
  }

  // ---- vs panel must be resident in LDS before P*V -----------------------
#ifdef HAVE_TDM
  if (threadIdx.x < 32) __builtin_amdgcn_s_wait_tensorcnt(0);
#else
  for (int e = threadIdx.x; e < S * DH; e += 256) vs_l[e] = vb[e];
#endif
  __syncthreads();

  // ---- Phase 3: heads(16xDH) = P(16xS) * vs(SxDH); K split across waves --
  const int n0 = (wave & 3) * 16;
  const int k0 = (wave >> 2) * (S / 2);
  floatx8 hacc = {};
  for (int kk = 0; kk < S / 2; kk += 32) {
    const int kg = k0 + kk;
    const __bf16* __restrict__ prow = p_bf + arow * S + kg;
    bf16x16 a, bm;
#pragma unroll
    for (int i = 0; i < 8; ++i) {        // two contiguous 16B runs -> b128
      a[i]     = prow[ak + i];
      a[8 + i] = prow[8 + ak + i];
    }
#pragma unroll
    for (int j = 0; j < 16; ++j)
      bm[j] = vs_l[(long)(kg + bk + j) * DH + n0 + bcol];
    hacc = wmma_bf16(a, bm, hacc);
  }
  __syncthreads();                       // P no longer needed; reuse LDS
  if (wave >= 4) {
    float* __restrict__ stg = sc + (wave - 4) * 256;
#pragma unroll
    for (int r = 0; r < 8; ++r) stg[lane * 8 + r] = hacc[r];
  }
  __syncthreads();
  if (wave < 4) {
    const float* __restrict__ stg = sc + wave * 256;
#pragma unroll
    for (int r = 0; r < 8; ++r) hacc[r] += stg[lane * 8 + r];
#pragma unroll
    for (int r = 0; r < 8; ++r) {
      const int m = r + cm;
      atomicAdd(&hsum[((long)b * S + s0 + m) * DH + n0 + cn], hacc[r]);
    }
  }
}

// ---------------------------------------------------------------------------
// Output GEMM: out[m,:D] = (hsum[m,:DH] / H) * Wo[:DH,:D]
// grid = ((B*S)/16, D/64), block = 128; Wo tiles staged transposed in LDS.
// ---------------------------------------------------------------------------
__global__ void out_kernel(const float* __restrict__ hsum,
                           const float* __restrict__ Wo,
                           float* __restrict__ out) {
  __shared__ __bf16 wt[64 * 32];          // transposed Wo tile: wt[n*32 + k]
  const int lane = threadIdx.x & 31;
  const int wave = threadIdx.x >> 5;
  const int m0    = blockIdx.x * 16;
  const int nbase = blockIdx.y * 64;

  const int arow = lane & 15;
  const int ak   = (lane & 16) ? 16 : 0;
  const int bcol = lane & 15;
  const int bk   = (lane & 16) ? 16 : 0;

  const float* __restrict__ Arow = hsum + (long)(m0 + arow) * DH;
  floatx8 acc = {};
#pragma unroll
  for (int kk = 0; kk < DH; kk += 32) {
    __syncthreads();
    for (int e = threadIdx.x; e < 32 * 64; e += 128) {
      const int k = e >> 6;
      const int n = e & 63;
      wt[n * 32 + k] = (__bf16)Wo[(long)(kk + k) * D + nbase + n];
    }
    __syncthreads();
    bf16x16 a, bm;
#pragma unroll
    for (int i = 0; i < 8; ++i) {
      a[i]     = (__bf16)(Arow[kk + ak + i]      * 0.125f);   // /H
      a[8 + i] = (__bf16)(Arow[kk + 8 + ak + i] * 0.125f);
    }
    const __bf16* __restrict__ wrow = wt + (wave * 16 + bcol) * 32;
#pragma unroll
    for (int j = 0; j < 16; ++j) bm[j] = wrow[bk + j];
    acc = wmma_bf16(a, bm, acc);
  }
#pragma unroll
  for (int r = 0; r < 8; ++r) {
    const int m = r + ((lane & 16) ? 8 : 0);
    out[(long)(m0 + m) * D + nbase + wave * 16 + (lane & 15)] = acc[r];
  }
}

// ---------------------------------------------------------------------------
extern "C" void kernel_launch(void* const* d_in, const int* in_sizes, int n_in,
                              void* d_out, int out_size, void* d_ws, size_t ws_size,
                              hipStream_t stream) {
  (void)in_sizes; (void)n_in; (void)out_size; (void)ws_size;

  const float* q  = (const float*)d_in[0];
  const float* k  = (const float*)d_in[1];
  const float* v  = (const float*)d_in[2];
  const unsigned char* mask = (const unsigned char*)d_in[3];
  const float* Wq = (const float*)d_in[4];
  const float* Wk = (const float*)d_in[5];
  const float* Wv = (const float*)d_in[6];
  const float* Wo = (const float*)d_in[7];

  float* out  = (float*)d_out;                   // [B,S,D]
  float* attn = out + (long)B * S * D;           // [B,H,S,S]

  constexpr long QH_ELEMS = (long)B * H * S * DH;
  constexpr long VS_ELEMS = (long)B * S * DH;
  __bf16* qh = (__bf16*)d_ws;
  __bf16* kh = qh + QH_ELEMS;
  __bf16* vs = kh + QH_ELEMS;
  float*  hsum = (float*)(vs + VS_ELEMS);        // [B,S,DH] f32 accumulator

  constexpr size_t ATTN_LDS = (size_t)16 * S * 4   // f32 scores
                            + (size_t)16 * S * 2   // bf16 P
                            + (size_t)S * DH * 2;  // bf16 vs panel

  zero_kernel<<<dim3((unsigned)(VS_ELEMS / 256)), 256, 0, stream>>>(hsum, VS_ELEMS);
  proj_kernel<<<dim3(S/16, B*H), 128, 0, stream>>>(q, Wq, qh, H, (long)D * DH);
  proj_kernel<<<dim3(S/16, B*H), 128, 0, stream>>>(k, Wk, kh, H, (long)D * DH);
  proj_kernel<<<dim3(S/16, B),   128, 0, stream>>>(v, Wv, vs, 1, 0);
  attn_kernel<<<dim3(S/16, B*H), 256, ATTN_LDS, stream>>>(qh, kh, vs, mask, attn, hsum);
  out_kernel<<<dim3((B*S)/16, D/64), 128, 0, stream>>>(hsum, Wo, out);
}
